// PAM_Module_84112639524977
// MI455X (gfx1250) — matrix operand order repaired and checked by reading, not compiled, not run on hardware
//
#include <hip/hip_runtime.h>
#include <hip/hip_bf16.h>

typedef __attribute__((ext_vector_type(16))) _Float16 v16h;
typedef __attribute__((ext_vector_type(8)))  _Float16 v8h;
typedef __attribute__((ext_vector_type(8)))  float    v8f;

#define BATCH 4
#define CCH   512
#define CQK   64
#define HW    4096

__device__ inline v8f wmma16(v16h a, v16h b, v8f c) {
    return __builtin_amdgcn_wmma_f32_16x16x32_f16(false, a, false, b, (short)0, c, false, false);
}
__device__ inline v8h  ld8h (const _Float16* p) { return *(const v8h*)p; }
__device__ inline v16h ld16h(const _Float16* p) { return *(const v16h*)p; }
__device__ inline v16h cat16(v8h lo, v8h hi) {
    v16h r;
#pragma unroll
    for (int i = 0; i < 8; ++i) { r[i] = lo[i]; r[i + 8] = hi[i]; }
    return r;
}
__device__ inline v8f zero8() {
    v8f z;
#pragma unroll
    for (int i = 0; i < 8; ++i) z[i] = 0.0f;
    return z;
}

// ---------------------------------------------------------------------------
// Kernel 1: 1x1-conv projection as WMMA GEMM: out[d,i] = sum_c W[d,c]*x[b,c,i]+bias[d]
// 8 waves/block, each wave owns one 16(M=chan) x 16(N=pixel) tile, K=Cin in steps of 32.
// TRANS=1 -> store out[b][i][d] f16 (for Q/K, one v8h store), else out[b][d][i] (V).
// ---------------------------------------------------------------------------
template <int TRANS>
__global__ void __launch_bounds__(256)
proj_kernel(const float* __restrict__ x, const float* __restrict__ W,
            const float* __restrict__ bias, _Float16* __restrict__ out, int Cout)
{
    const int b    = blockIdx.z;
    const int m0   = blockIdx.y << 4;                 // output-channel tile
    const int wave = threadIdx.x >> 5;
    const int lane = threadIdx.x & 31;
    const int hg   = lane >> 4;                       // half-group
    const int ln   = lane & 15;
    const int i0   = (blockIdx.x << 7) + (wave << 4); // pixel tile

    const float* xb = x + (size_t)b * CCH * HW;
    v8f acc = zero8();

    for (int kk = 0; kk < CCH; kk += 32) {
        // A-frag: W[m][k], m = m0+ln, k in {kk+hg*8..+8, kk+16+hg*8..+8}; 32B-aligned.
        const float* wr = W + (size_t)(m0 + ln) * CCH + kk + hg * 8;
        const float4 w0 = *(const float4*)(wr);
        const float4 w1 = *(const float4*)(wr + 4);
        const float4 w2 = *(const float4*)(wr + 16);
        const float4 w3 = *(const float4*)(wr + 20);
        v16h a;
        a[0]  = (_Float16)w0.x; a[1]  = (_Float16)w0.y; a[2]  = (_Float16)w0.z; a[3]  = (_Float16)w0.w;
        a[4]  = (_Float16)w1.x; a[5]  = (_Float16)w1.y; a[6]  = (_Float16)w1.z; a[7]  = (_Float16)w1.w;
        a[8]  = (_Float16)w2.x; a[9]  = (_Float16)w2.y; a[10] = (_Float16)w2.z; a[11] = (_Float16)w2.w;
        a[12] = (_Float16)w3.x; a[13] = (_Float16)w3.y; a[14] = (_Float16)w3.z; a[15] = (_Float16)w3.w;
        // B-frag: x[k][n], n = i0+ln, k = kk + hg*8 + h + (h&8) (stride HW in memory)
        const float* xc = xb + (size_t)(kk + hg * 8) * HW + i0 + ln;
        v16h bm;
#pragma unroll
        for (int h = 0; h < 16; ++h) bm[h] = (_Float16)xc[(size_t)(h + (h & 8)) * HW];
        acc = wmma16(a, bm, acc);
    }

    const int n = i0 + ln;  // pixel
    if (TRANS) {
        v8h vals;
#pragma unroll
        for (int r = 0; r < 8; ++r)
            vals[r] = (_Float16)(acc[r] + bias[m0 + r + hg * 8]);
        *(v8h*)(out + ((size_t)b * HW + n) * CQK + m0 + hg * 8) = vals;
    } else {
#pragma unroll
        for (int r = 0; r < 8; ++r) {
            int m = m0 + r + hg * 8;
            out[((size_t)b * Cout + m) * HW + n] = (_Float16)(acc[r] + bias[m]);
        }
    }
}

// ---------------------------------------------------------------------------
// Kernel 2: softmax row stats. Block = 16 i-rows; 8 waves sweep disjoint j-tiles,
// S = Q^T K via WMMA. Online softmax is kept PER-LANE (each lane owns one column
// of the D-frag), so the hot loop is pure VALU (exp/max/fma) with no shuffles.
// A single cross-lane (m,l) merge + LDS cross-wave combine runs once at the end.
// ---------------------------------------------------------------------------
__global__ void __launch_bounds__(256)
softmax_stats_kernel(const _Float16* __restrict__ qT, const _Float16* __restrict__ kT,
                     float* __restrict__ mG, float* __restrict__ lG)
{
    __shared__ float sm[8][16];
    __shared__ float sl[8][16];

    const int b    = blockIdx.y;
    const int i0   = blockIdx.x << 4;
    const int wave = threadIdx.x >> 5;
    const int lane = threadIdx.x & 31;
    const int hg   = lane >> 4;
    const int ln   = lane & 15;

    const _Float16* qrow = qT + ((size_t)b * HW + i0 + ln) * CQK + hg * 8;
    v16h aq0 = cat16(ld8h(qrow),      ld8h(qrow + 16));
    v16h aq1 = cat16(ld8h(qrow + 32), ld8h(qrow + 48));

    float mrun[8], lrun[8];
#pragma unroll
    for (int r = 0; r < 8; ++r) { mrun[r] = -3.0e38f; lrun[r] = 0.0f; }

    for (int jt = wave; jt < HW / 16; jt += 8) {
        const _Float16* kcol = kT + ((size_t)b * HW + (jt << 4) + ln) * CQK + hg * 8;
        v16h b0 = cat16(ld8h(kcol),      ld8h(kcol + 16));
        v16h b1 = cat16(ld8h(kcol + 32), ld8h(kcol + 48));
        v8f s = zero8();
        s = wmma16(aq0, b0, s);
        s = wmma16(aq1, b1, s);
        // per-lane exact online softmax over this lane's column elements
#pragma unroll
        for (int r = 0; r < 8; ++r) {
            float v  = s[r];
            float nm = fmaxf(mrun[r], v);
            lrun[r]  = lrun[r] * __expf(mrun[r] - nm) + __expf(v - nm);
            mrun[r]  = nm;
        }
    }

    // one cross-lane merge of (m,l) pairs within each 16-lane group
#pragma unroll
    for (int r = 0; r < 8; ++r) {
#pragma unroll
        for (int d = 1; d < 16; d <<= 1) {
            float om = __shfl_xor(mrun[r], d, 32);
            float ol = __shfl_xor(lrun[r], d, 32);
            float nm = fmaxf(mrun[r], om);
            lrun[r]  = lrun[r] * __expf(mrun[r] - nm) + ol * __expf(om - nm);
            mrun[r]  = nm;
        }
    }

    if (ln == 0) {
#pragma unroll
        for (int r = 0; r < 8; ++r) {
            sm[wave][r + hg * 8] = mrun[r];
            sl[wave][r + hg * 8] = lrun[r];
        }
    }
    __syncthreads();
    if (threadIdx.x < 16) {
        int row = threadIdx.x;
        float M = -3.0e38f;
#pragma unroll
        for (int w = 0; w < 8; ++w) M = fmaxf(M, sm[w][row]);
        float L = 0.0f;
#pragma unroll
        for (int w = 0; w < 8; ++w) L += sl[w][row] * __expf(sm[w][row] - M);
        mG[(size_t)b * HW + i0 + row] = M;
        lG[(size_t)b * HW + i0 + row] = L;
    }
}

// ---------------------------------------------------------------------------
// Kernel 3: out = gamma * (V @ softmax(Q^T K)^T) + x.
// Block = 32 i-rows x all 512 channels (2 row-tiles to double V reuse).
// Per 128-wide j-slab: each of 8 waves computes two 16x16 S tiles (4 WMMA),
// normalizes -> P f16 in LDS; barrier; every wave multiplies both P row-tiles
// (A-frags from LDS) by its 64-channel V slice (32 WMMA, V loaded once).
// ---------------------------------------------------------------------------
__global__ void __launch_bounds__(256)
attn_out_kernel(const _Float16* __restrict__ qT, const _Float16* __restrict__ kT,
                const _Float16* __restrict__ vH, const float* __restrict__ mG,
                const float* __restrict__ lG, const float* __restrict__ x,
                const float* __restrict__ gamma, float* __restrict__ out)
{
    __shared__ _Float16 P[32][136];  // 32x128 + pad(8) keeps rows 16B-aligned

    const int b    = blockIdx.y;
    const int i0   = blockIdx.x << 5;   // 32 rows per block
    const int wave = threadIdx.x >> 5;
    const int lane = threadIdx.x & 31;
    const int hg   = lane >> 4;
    const int ln   = lane & 15;
    const int c0   = wave << 6;         // 64 channels per wave

    // Q A-frags for both row-tiles (same for every wave)
    const _Float16* qrowA = qT + ((size_t)b * HW + i0 + ln) * CQK + hg * 8;
    const _Float16* qrowB = qrowA + (size_t)16 * CQK;
    v16h aq0a = cat16(ld8h(qrowA),      ld8h(qrowA + 16));
    v16h aq1a = cat16(ld8h(qrowA + 32), ld8h(qrowA + 48));
    v16h aq0b = cat16(ld8h(qrowB),      ld8h(qrowB + 16));
    v16h aq1b = cat16(ld8h(qrowB + 32), ld8h(qrowB + 48));

    // softmax stats for this lane's rows (D-frag rows m = r + hg*8, per tile)
    float mrow[2][8], invl[2][8];
#pragma unroll
    for (int t = 0; t < 2; ++t)
#pragma unroll
        for (int r = 0; r < 8; ++r) {
            int m = t * 16 + r + hg * 8;
            mrow[t][r] = mG[(size_t)b * HW + i0 + m];
            invl[t][r] = 1.0f / lG[(size_t)b * HW + i0 + m];
        }

    v8f o[2][4];
#pragma unroll
    for (int t = 0; t < 2; ++t)
#pragma unroll
        for (int nt = 0; nt < 4; ++nt) o[t][nt] = zero8();

    for (int j0 = 0; j0 < HW; j0 += 128) {
        // --- S tiles for this wave's 16-j strip, both row tiles ---
        const _Float16* kcol = kT + ((size_t)b * HW + j0 + (wave << 4) + ln) * CQK + hg * 8;
        v16h kb0 = cat16(ld8h(kcol),      ld8h(kcol + 16));
        v16h kb1 = cat16(ld8h(kcol + 32), ld8h(kcol + 48));
        v8f sA = zero8(), sB = zero8();
        sA = wmma16(aq0a, kb0, sA);
        sA = wmma16(aq1a, kb1, sA);
        sB = wmma16(aq0b, kb0, sB);
        sB = wmma16(aq1b, kb1, sB);

        __syncthreads();   // previous slab's P fully consumed
#pragma unroll
        for (int r = 0; r < 8; ++r) {
            int m = r + hg * 8;
            P[m][(wave << 4) + ln]      = (_Float16)(__expf(sA[r] - mrow[0][r]) * invl[0][r]);
            P[m + 16][(wave << 4) + ln] = (_Float16)(__expf(sB[r] - mrow[1][r]) * invl[1][r]);
        }
        __syncthreads();   // P slab ready

        // --- PV: O[i-tiles, c0..c0+63] += P * V^T ---
#pragma unroll
        for (int kc = 0; kc < 4; ++kc) {
            const _Float16* prowA = &P[ln][kc * 32 + hg * 8];
            const _Float16* prowB = &P[ln + 16][kc * 32 + hg * 8];
            v16h paA = cat16(ld8h(prowA), ld8h(prowA + 16));
            v16h paB = cat16(ld8h(prowB), ld8h(prowB + 16));
            const _Float16* vb = vH + ((size_t)b * CCH + c0 + ln) * HW + j0 + kc * 32 + hg * 8;
            if (kc == 0) {
#pragma unroll
                for (int nt = 0; nt < 4; ++nt)
                    __builtin_prefetch(vb + (size_t)(nt * 16) * HW + 128, 0, 3);
            }
#pragma unroll
            for (int nt = 0; nt < 4; ++nt) {
                v16h vbm = cat16(ld8h(vb + (size_t)(nt * 16) * HW), ld8h(vb + (size_t)(nt * 16) * HW + 16));
                o[0][nt] = wmma16(paA, vbm, o[0][nt]);
                o[1][nt] = wmma16(paB, vbm, o[1][nt]);
            }
        }
    }

    const float g = gamma[0];
#pragma unroll
    for (int t = 0; t < 2; ++t)
#pragma unroll
        for (int nt = 0; nt < 4; ++nt) {
            int c = c0 + nt * 16 + ln;
            int i = i0 + t * 16 + hg * 8;          // rows r..r+3 consecutive
            size_t idx = ((size_t)b * CCH + c) * HW + i;
            const float4 x0 = *(const float4*)(x + idx);
            const float4 x1 = *(const float4*)(x + idx + 4);
            float4 r0, r1;
            r0.x = g * o[t][nt][0] + x0.x; r0.y = g * o[t][nt][1] + x0.y;
            r0.z = g * o[t][nt][2] + x0.z; r0.w = g * o[t][nt][3] + x0.w;
            r1.x = g * o[t][nt][4] + x1.x; r1.y = g * o[t][nt][5] + x1.y;
            r1.z = g * o[t][nt][6] + x1.z; r1.w = g * o[t][nt][7] + x1.w;
            *(float4*)(out + idx)     = r0;
            *(float4*)(out + idx + 4) = r1;
        }
}

// ---------------------------------------------------------------------------
extern "C" void kernel_launch(void* const* d_in, const int* in_sizes, int n_in,
                              void* d_out, int out_size, void* d_ws, size_t ws_size,
                              hipStream_t stream)
{
    const float* x     = (const float*)d_in[0];
    const float* Wq    = (const float*)d_in[1];
    const float* bq    = (const float*)d_in[2];
    const float* Wk    = (const float*)d_in[3];
    const float* bk    = (const float*)d_in[4];
    const float* Wv    = (const float*)d_in[5];
    const float* bv    = (const float*)d_in[6];
    const float* gamma = (const float*)d_in[7];
    float* out = (float*)d_out;

    // workspace: qT(2MB) kT(2MB) vH(16MB) m(64KB) l(64KB)  ~= 20.1 MB
    _Float16* qT = (_Float16*)d_ws;
    _Float16* kT = qT + (size_t)BATCH * HW * CQK;
    _Float16* vH = kT + (size_t)BATCH * HW * CQK;
    float*    mG = (float*)(vH + (size_t)BATCH * CCH * HW);
    float*    lG = mG + (size_t)BATCH * HW;

    dim3 blk(256);
    proj_kernel<1><<<dim3(HW / 128, CQK / 16, BATCH), blk, 0, stream>>>(x, Wq, bq, qT, CQK);
    proj_kernel<1><<<dim3(HW / 128, CQK / 16, BATCH), blk, 0, stream>>>(x, Wk, bk, kT, CQK);
    proj_kernel<0><<<dim3(HW / 128, CCH / 16, BATCH), blk, 0, stream>>>(x, Wv, bv, vH, CCH);
    softmax_stats_kernel<<<dim3(HW / 16, BATCH), blk, 0, stream>>>(qT, kT, mG, lG);
    attn_out_kernel<<<dim3(HW / 32, BATCH), blk, 0, stream>>>(qT, kT, vH, mG, lG, x, gamma, out);
}
